// Decoder_41747082117284
// MI455X (gfx1250) — matrix operand order repaired and checked by reading, not compiled, not run on hardware
//
#include <hip/hip_runtime.h>
#include <hip/hip_bf16.h>

// ---------------------------------------------------------------------------
// Types for CDNA5 WMMA (wave32)
// ---------------------------------------------------------------------------
typedef __bf16 bf16_t;
typedef __attribute__((ext_vector_type(16))) __bf16 v16bf;
typedef __attribute__((ext_vector_type(8)))  __bf16 v8bf;
typedef __attribute__((ext_vector_type(8)))  float  v8f;
typedef __attribute__((ext_vector_type(4)))  float  f32x4;

__device__ __forceinline__ v8f wmma_bf16(v16bf a, v16bf b, v8f c) {
  // D = A(16x32 bf16) * B(32x16 bf16) + C(16x16 f32)
  return __builtin_amdgcn_wmma_f32_16x16x32_bf16(
      /*neg_a=*/false, a, /*neg_b=*/false, b,
      /*c_mod=*/(short)0, c, /*reuse_a=*/false, /*reuse_b=*/false);
}

// A-fragment: per-lane elements are 16 consecutive k
// (k = kA + {0..15}).  Two 16B ds_load_b128.
__device__ __forceinline__ v16bf ld_frag_a(const bf16_t* p) {
  v8bf lo = *(const v8bf*)p;
  v8bf hi = *(const v8bf*)(p + 8);
  return __builtin_shufflevector(lo, hi, 0, 1, 2, 3, 4, 5, 6, 7,
                                 8, 9, 10, 11, 12, 13, 14, 15);
}
// B-fragment from a k-contiguous (transposed) panel: 16 consecutive elems.
__device__ __forceinline__ v16bf ld_frag_b(const bf16_t* p) {
  v8bf lo = *(const v8bf*)p;
  v8bf hi = *(const v8bf*)(p + 8);
  return __builtin_shufflevector(lo, hi, 0, 1, 2, 3, 4, 5, 6, 7,
                                 8, 9, 10, 11, 12, 13, 14, 15);
}

__device__ __forceinline__ v8bf pack8(f32x4 a, f32x4 b) {
  v8bf r;
  r[0] = (bf16_t)a.x; r[1] = (bf16_t)a.y; r[2] = (bf16_t)a.z; r[3] = (bf16_t)a.w;
  r[4] = (bf16_t)b.x; r[5] = (bf16_t)b.y; r[6] = (bf16_t)b.z; r[7] = (bf16_t)b.w;
  return r;
}

__device__ __forceinline__ float halfmax16(float v) {
  v = fmaxf(v, __shfl_xor(v, 1, 32));
  v = fmaxf(v, __shfl_xor(v, 2, 32));
  v = fmaxf(v, __shfl_xor(v, 4, 32));
  v = fmaxf(v, __shfl_xor(v, 8, 32));
  return v;
}
__device__ __forceinline__ float halfsum16(float v) {
  v += __shfl_xor(v, 1, 32);
  v += __shfl_xor(v, 2, 32);
  v += __shfl_xor(v, 4, 32);
  v += __shfl_xor(v, 8, 32);
  return v;
}

// ---------------------------------------------------------------------------
// GEMM: C[M,N] = op(A[M,K] @ B[K,N]); fp32 in/out, bf16 WMMA inside.
// Tile 128x64, BK=64, 256 threads = 8 waves; each wave owns a 32x32 patch
// (2x2 subtiles -> 4 WMMAs per k-step, 8 per K-tile).
// B is staged TRANSPOSED in LDS so B-fragments are k-contiguous b128 loads.
// ---------------------------------------------------------------------------
template <int RELU>
__global__ __launch_bounds__(256)
void gemm_bf16(const float* __restrict__ A, const float* __restrict__ B,
               float* __restrict__ C, int M, int N, int K) {
  constexpr int BM = 128, BN = 64, BK = 64, LDP = BK + 8;
  __shared__ __align__(16) bf16_t sA[BM][LDP];    // [m][k]
  __shared__ __align__(16) bf16_t sBt[BN][LDP];   // [n][k]  (transposed)

  const int tid  = threadIdx.x;
  const int lane = tid & 31;
  const int wave = tid >> 5;
  const int mBase = blockIdx.y * BM;
  const int nBase = blockIdx.x * BN;
  const int mOff  = (wave & 3) * 32;
  const int nOff  = (wave >> 2) * 32;

  // fragment lane mapping (this code's contiguous k order: k = 16*(lane>>4) + s, same for A and B)
  const int mA = lane & 15;
  const int kA = (lane >> 4) * 16;
  const int nB = lane & 15;
  const int kB = (lane >> 4) * 16;

  // B staging pattern: thread covers one n-column, 16 consecutive k.
  const int bN = tid & 63;
  const int bK = (tid >> 6) * 16;

  v8f acc00 = {}, acc01 = {}, acc10 = {}, acc11 = {};

  for (int kt = 0; kt < K; kt += BK) {
    // ---- stage A (128x64): 1024 8-elem chunks, 4/thread, b128 LDS stores --
#pragma unroll
    for (int i = 0; i < 4; ++i) {
      int idx = tid + i * 256;
      int r   = idx >> 3;
      int c8  = (idx & 7) * 8;
      const float* ap = A + (size_t)(mBase + r) * K + kt + c8;
      f32x4 v0 = *(const f32x4*)ap;
      f32x4 v1 = *(const f32x4*)(ap + 4);
      *(v8bf*)&sA[r][c8] = pack8(v0, v1);
    }
    // ---- stage B transposed (k fast axis): coalesced dword loads ----------
    {
      float tmp[16];
#pragma unroll
      for (int i = 0; i < 16; ++i)
        tmp[i] = B[(size_t)(kt + bK + i) * N + nBase + bN];
      v8bf p0, p1;
#pragma unroll
      for (int i = 0; i < 8; ++i) { p0[i] = (bf16_t)tmp[i]; p1[i] = (bf16_t)tmp[8 + i]; }
      *(v8bf*)&sBt[bN][bK]     = p0;
      *(v8bf*)&sBt[bN][bK + 8] = p1;
    }
    __syncthreads();

#pragma unroll
    for (int s = 0; s < 2; ++s) {
      v16bf a0 = ld_frag_a(&sA[mOff + mA][s * 32 + kA]);
      v16bf a1 = ld_frag_a(&sA[mOff + 16 + mA][s * 32 + kA]);
      v16bf b0 = ld_frag_b(&sBt[nOff + nB][s * 32 + kB]);
      v16bf b1 = ld_frag_b(&sBt[nOff + 16 + nB][s * 32 + kB]);
      acc00 = wmma_bf16(a0, b0, acc00);
      acc01 = wmma_bf16(a0, b1, acc01);
      acc10 = wmma_bf16(a1, b0, acc10);
      acc11 = wmma_bf16(a1, b1, acc11);
    }
    __syncthreads();
  }

  // ---- epilogue: C/D layout row = (lane>>4)*8 + r, col = lane&15 ----------
  const int row = mBase + mOff + (lane >> 4) * 8;
  const int col = nBase + nOff + (lane & 15);
#pragma unroll
  for (int r = 0; r < 8; ++r) {
    float x00 = acc00[r], x01 = acc01[r], x10 = acc10[r], x11 = acc11[r];
    if (RELU) {
      x00 = fmaxf(x00, 0.f); x01 = fmaxf(x01, 0.f);
      x10 = fmaxf(x10, 0.f); x11 = fmaxf(x11, 0.f);
    }
    C[(size_t)(row + r) * N + col]           = x00;
    C[(size_t)(row + r) * N + col + 16]      = x01;
    C[(size_t)(row + 16 + r) * N + col]      = x10;
    C[(size_t)(row + 16 + r) * N + col + 16] = x11;
  }
}

// ---------------------------------------------------------------------------
// Flash attention. Q,K,V,O are [B*rows, 1024] fp32, head h at cols h*64.
// Grid (T/128, H, B). 256 threads = 8 waves; wave w owns 16 q-rows.
// sK kept row-major [key][d] (scores' B operand is K^T -> k-dim is d),
// sVt transposed [d][key] (P@V's k-dim is key).  All fragments are b128.
// ---------------------------------------------------------------------------
__global__ __launch_bounds__(256)
void attention(const float* __restrict__ Qm, const float* __restrict__ Km,
               const float* __restrict__ Vm, float* __restrict__ Om,
               int T, int S, int ld, int causal) {
  constexpr int QT = 128, KT = 64, D = 64, LDP = 72;
  __shared__ __align__(16) bf16_t sQ[QT][LDP];        // [q][d], used once
  __shared__ __align__(16) bf16_t sK[KT][LDP];        // [key][d]
  __shared__ __align__(16) bf16_t sVt[D][LDP];        // [d][key]
  __shared__ __align__(16) bf16_t sP[8][16][LDP];     // per-wave P panel

  const int tid = threadIdx.x, lane = tid & 31, wave = tid >> 5;
  const int b = blockIdx.z, h = blockIdx.y;
  const int qBase = blockIdx.x * QT;
  const int hOff  = h * D;

  const int mA = lane & 15;
  const int kA = (lane >> 4) * 16;
  const int nB = lane & 15;
  const int kB = (lane >> 4) * 16;

  // ---- stage Q (scaled by 1/sqrt(DK)) and pick up A-fragments -------------
#pragma unroll
  for (int i = 0; i < 4; ++i) {
    int idx = tid + i * 256;
    int r   = idx >> 3;
    int c8  = (idx & 7) * 8;
    const float* qp = Qm + (size_t)(b * T + qBase + r) * ld + hOff + c8;
    f32x4 v0 = *(const f32x4*)qp;
    f32x4 v1 = *(const f32x4*)(qp + 4);
    v0.x *= 0.125f; v0.y *= 0.125f; v0.z *= 0.125f; v0.w *= 0.125f;
    v1.x *= 0.125f; v1.y *= 0.125f; v1.z *= 0.125f; v1.w *= 0.125f;
    *(v8bf*)&sQ[r][c8] = pack8(v0, v1);
  }
  __syncthreads();
  v16bf aQ0 = ld_frag_a(&sQ[wave * 16 + mA][kA]);
  v16bf aQ1 = ld_frag_a(&sQ[wave * 16 + mA][32 + kA]);

  v8f accO[4] = {{}, {}, {}, {}};
  float mrow[8], lrow[8];
#pragma unroll
  for (int r = 0; r < 8; ++r) { mrow[r] = -1e30f; lrow[r] = 0.f; }

  const int rBase   = (lane >> 4) * 8;
  const int myQrow0 = qBase + wave * 16 + rBase;
  const int kEnd    = causal ? ((qBase + QT < S) ? qBase + QT : S) : S;

  // V staging pattern: thread covers one d-column, 16 consecutive keys.
  const int vD = tid & 63;
  const int vK = (tid >> 6) * 16;

  for (int kb = 0; kb < kEnd; kb += KT) {
    // ---- stage K row-major: 4 float4 loads + 2 b128 stores per thread ----
    {
      int kk = tid >> 2;
      int d0 = (tid & 3) * 16;
      const float* kp = Km + (size_t)(b * S + kb + kk) * ld + hOff + d0;
      f32x4 k0 = *(const f32x4*)kp;
      f32x4 k1 = *(const f32x4*)(kp + 4);
      f32x4 k2 = *(const f32x4*)(kp + 8);
      f32x4 k3 = *(const f32x4*)(kp + 12);
      *(v8bf*)&sK[kk][d0]     = pack8(k0, k1);
      *(v8bf*)&sK[kk][d0 + 8] = pack8(k2, k3);
    }
    // ---- stage V transposed: coalesced dword loads along d ---------------
    {
      float tmp[16];
#pragma unroll
      for (int i = 0; i < 16; ++i)
        tmp[i] = Vm[(size_t)(b * S + kb + vK + i) * ld + hOff + vD];
      v8bf p0, p1;
#pragma unroll
      for (int i = 0; i < 8; ++i) { p0[i] = (bf16_t)tmp[i]; p1[i] = (bf16_t)tmp[8 + i]; }
      *(v8bf*)&sVt[vD][vK]     = p0;
      *(v8bf*)&sVt[vD][vK + 8] = p1;
    }
    __syncthreads();

    // ---- scores S = Qs @ K^T : 4 subtiles x 2 k-steps --------------------
    v8f accS[4] = {{}, {}, {}, {}};
#pragma unroll
    for (int t = 0; t < 4; ++t) {
      v16bf b0 = ld_frag_b(&sK[t * 16 + nB][kB]);        // k-dim = d: 0..31
      v16bf b1 = ld_frag_b(&sK[t * 16 + nB][32 + kB]);   // d: 32..63
      accS[t] = wmma_bf16(aQ0, b0, accS[t]);
      accS[t] = wmma_bf16(aQ1, b1, accS[t]);
    }

    // ---- causal mask + online softmax ------------------------------------
#pragma unroll
    for (int r = 0; r < 8; ++r) {
      int grow = myQrow0 + r;
      if (causal) {
#pragma unroll
        for (int t = 0; t < 4; ++t) {
          int gcol = kb + t * 16 + nB;
          if (gcol > grow) accS[t][r] = -1e9f;
        }
      }
      float bm = fmaxf(fmaxf(accS[0][r], accS[1][r]),
                       fmaxf(accS[2][r], accS[3][r]));
      bm = halfmax16(bm);
      float mnew  = fmaxf(mrow[r], bm);
      float alpha = __expf(mrow[r] - mnew);
      mrow[r] = mnew;
      float ps = 0.f;
#pragma unroll
      for (int t = 0; t < 4; ++t) {
        float p = __expf(accS[t][r] - mnew);
        accS[t][r] = p;
        ps += p;
      }
      ps = halfsum16(ps);
      lrow[r] = lrow[r] * alpha + ps;
#pragma unroll
      for (int t = 0; t < 4; ++t) accO[t][r] *= alpha;
#pragma unroll
      for (int t = 0; t < 4; ++t)          // cross-lane scatter (unavoidable)
        sP[wave][rBase + r][t * 16 + nB] = (bf16_t)accS[t][r];
    }

    // ---- O += P @ V ------------------------------------------------------
    v16bf aP0 = ld_frag_a(&sP[wave][mA][kA]);
    v16bf aP1 = ld_frag_a(&sP[wave][mA][32 + kA]);
#pragma unroll
    for (int t = 0; t < 4; ++t) {
      v16bf b0 = ld_frag_b(&sVt[t * 16 + nB][kB]);       // k-dim = key: 0..31
      v16bf b1 = ld_frag_b(&sVt[t * 16 + nB][32 + kB]);  // key: 32..63
      accO[t] = wmma_bf16(aP0, b0, accO[t]);
      accO[t] = wmma_bf16(aP1, b1, accO[t]);
    }
    __syncthreads();
  }

  // ---- finalize: O / l --------------------------------------------------
#pragma unroll
  for (int r = 0; r < 8; ++r) {
    float inv = 1.f / lrow[r];
    size_t rowOff = (size_t)(b * T + myQrow0 + r) * ld + hOff;
#pragma unroll
    for (int t = 0; t < 4; ++t)
      Om[rowOff + t * 16 + nB] = accO[t][r] * inv;
  }
}

// ---------------------------------------------------------------------------
// Y = LayerNorm(X + R), one block per row (DMc = 1024), 256 threads.
// ---------------------------------------------------------------------------
__global__ __launch_bounds__(256)
void add_layernorm(const float* __restrict__ X, const float* __restrict__ R,
                   float* __restrict__ Y, int DMc) {
  __shared__ float red[2][8];
  const int row = blockIdx.x;
  const float* xp = X + (size_t)row * DMc;
  const float* rp = R + (size_t)row * DMc;
  float v[4], s = 0.f, ss = 0.f;
#pragma unroll
  for (int i = 0; i < 4; ++i) {
    int c = threadIdx.x + i * 256;
    v[i] = xp[c] + rp[c];
    s += v[i];
    ss += v[i] * v[i];
  }
#pragma unroll
  for (int m = 16; m >= 1; m >>= 1) {
    s  += __shfl_xor(s, m, 32);
    ss += __shfl_xor(ss, m, 32);
  }
  const int wave = threadIdx.x >> 5, lane = threadIdx.x & 31;
  if (lane == 0) { red[0][wave] = s; red[1][wave] = ss; }
  __syncthreads();
  if (wave == 0) {
    float a  = (lane < 8) ? red[0][lane] : 0.f;
    float b2 = (lane < 8) ? red[1][lane] : 0.f;
#pragma unroll
    for (int m = 4; m >= 1; m >>= 1) {
      a  += __shfl_xor(a, m, 32);
      b2 += __shfl_xor(b2, m, 32);
    }
    if (lane == 0) { red[0][0] = a; red[1][0] = b2; }
  }
  __syncthreads();
  float mean = red[0][0] / DMc;
  float var  = red[1][0] / DMc - mean * mean;
  float rstd = rsqrtf(var + 1e-5f);
#pragma unroll
  for (int i = 0; i < 4; ++i) {
    int c = threadIdx.x + i * 256;
    Y[(size_t)row * DMc + c] = (v[i] - mean) * rstd;
  }
}

// ---------------------------------------------------------------------------
// Host orchestration
// ---------------------------------------------------------------------------
extern "C" void kernel_launch(void* const* d_in, const int* in_sizes, int n_in,
                              void* d_out, int out_size, void* d_ws, size_t ws_size,
                              hipStream_t stream) {
  (void)in_sizes; (void)n_in; (void)out_size; (void)ws_size;
  constexpr int L = 4, DM = 1024, H = 16, DK = 64, DFF = 4096;
  constexpr int Bsz = 4, T = 1024, S = 1024;
  constexpr int M = Bsz * T;               // 4096 rows
  constexpr size_t NT = (size_t)M * DM;    // 4M floats per activation buffer

  const float* dec = (const float*)d_in[0];
  const float* enc = (const float*)d_in[1];
  // d_in[2], d_in[3]: masks (semantics known: causal self / unmasked cross)
  const float* sWq = (const float*)d_in[4];
  const float* sWk = (const float*)d_in[5];
  const float* sWv = (const float*)d_in[6];
  const float* sWo = (const float*)d_in[7];
  const float* cWq = (const float*)d_in[8];
  const float* cWk = (const float*)d_in[9];
  const float* cWv = (const float*)d_in[10];
  const float* cWo = (const float*)d_in[11];
  const float* W1  = (const float*)d_in[12];
  const float* W2  = (const float*)d_in[13];

  float* x   = (float*)d_ws;
  float* bq  = x  + NT;
  float* bk  = bq + NT;
  float* bv  = bk + NT;
  float* bc  = bv + NT;          // attention context
  float* bo  = bc + NT;          // projection / FFN output
  float* hid = bq;               // FFN hidden aliases q|k|v|ctx (stages serialized)

  hipMemcpyAsync(x, dec, NT * sizeof(float), hipMemcpyDeviceToDevice, stream);

  const dim3 blk(256);
  const dim3 gP(DM / 64, M / 128);       // projections: N=1024
  const dim3 gF1(DFF / 64, M / 128);     // FFN up: N=4096
  const dim3 gAttn(T / 128, H, Bsz);
  const dim3 gLN(M);

  for (int l = 0; l < L; ++l) {
    const size_t wOff  = (size_t)l * DM * (H * DK);
    const size_t f1Off = (size_t)l * DM * DFF;
    const size_t f2Off = (size_t)l * DFF * DM;

    // ---------------- self attention (causal) ----------------
    gemm_bf16<0><<<gP, blk, 0, stream>>>(x, sWq + wOff, bq, M, DM, DM);
    gemm_bf16<0><<<gP, blk, 0, stream>>>(x, sWk + wOff, bk, M, DM, DM);
    gemm_bf16<0><<<gP, blk, 0, stream>>>(x, sWv + wOff, bv, M, DM, DM);
    attention<<<gAttn, blk, 0, stream>>>(bq, bk, bv, bc, T, T, DM, /*causal=*/1);
    gemm_bf16<0><<<gP, blk, 0, stream>>>(bc, sWo + wOff, bo, M, DM, DM);
    add_layernorm<<<gLN, blk, 0, stream>>>(bo, x, x, DM);

    // ---------------- cross attention (no mask) ----------------
    gemm_bf16<0><<<gP, blk, 0, stream>>>(x,   cWq + wOff, bq, M, DM, DM);
    gemm_bf16<0><<<gP, blk, 0, stream>>>(enc, cWk + wOff, bk, M, DM, DM);
    gemm_bf16<0><<<gP, blk, 0, stream>>>(enc, cWv + wOff, bv, M, DM, DM);
    attention<<<gAttn, blk, 0, stream>>>(bq, bk, bv, bc, T, S, DM, /*causal=*/0);
    gemm_bf16<0><<<gP, blk, 0, stream>>>(bc, cWo + wOff, bo, M, DM, DM);
    add_layernorm<<<gLN, blk, 0, stream>>>(bo, x, x, DM);

    // ---------------- FFN ----------------
    gemm_bf16<1><<<gF1, blk, 0, stream>>>(x, W1 + f1Off, hid, M, DFF, DM);
    gemm_bf16<0><<<gP,  blk, 0, stream>>>(hid, W2 + f2Off, bo, M, DM, DFF);
    float* ydst = (l == L - 1) ? (float*)d_out : x;
    add_layernorm<<<gLN, blk, 0, stream>>>(bo, x, ydst, DM);
  }
}
